// FSAS_89395449299478
// MI455X (gfx1250) — matrix operand order repaired and checked by reading, not compiled, not run on hardware
//
#include <hip/hip_runtime.h>
#include <hip/hip_bf16.h>

typedef __attribute__((ext_vector_type(16))) _Float16 v16h;
typedef __attribute__((ext_vector_type(8)))  _Float16 v8h;
typedef __attribute__((ext_vector_type(8)))  float    v8f;
typedef __attribute__((ext_vector_type(4)))  float    v4f;

#define DIM   64
#define C2    128
#define HH    256
#define WW    256
#define TILE  16
#define HALO  18
#define HP    324          // 18*18 halo pixels
#define XS_ROWS 336
#define XS_STRIDE 72       // halves: 64 + 8 pad -> 144B rows, bank-spread
#define WH_STRIDE 72       // halves
#define HS_STRIDE 344      // halves: 336 (padded col-tiles) + 8 -> stores never guarded
#define QKV_STRIDE 264     // floats (qs/ks are f32 now)
#define VS_STRIDE 264      // halves
#define TS_STRIDE 40       // halves
#define WO_STRIDE 40       // halves

// byte offsets into dynamic LDS; every section size is a multiple of 16B
#define B_XS   0
#define B_WHS  (B_XS   + XS_ROWS*XS_STRIDE*2)   // 48384
#define B_HS   (B_WHS  + 96*WH_STRIDE*2)        // +13824
#define B_QS   (B_HS   + 96*HS_STRIDE*2)        // +66048
#define B_KS   (B_QS   + 32*QKV_STRIDE*4)       // +33792
#define B_VS   (B_KS   + 32*QKV_STRIDE*4)       // +33792
#define B_TS   (B_VS   + 32*VS_STRIDE*2)        // +16896
#define B_WOS  (B_TS   + 256*TS_STRIDE*2)       // +20480
#define B_WDWS (B_WOS  + 64*WO_STRIDE*2)        // +5120
#define B_BDWS (B_WDWS + 96*12*4)               // +4608
#define B_BHS  (B_BDWS + 96*4)
#define SMEM_BYTES (B_BHS + 96*4)               // 243,712 B < 320KB WGP LDS

__global__ __launch_bounds__(256)
void fsas_fused(const float* __restrict__ x,   const float* __restrict__ wh,
                const float* __restrict__ bh,  const float* __restrict__ wdw,
                const float* __restrict__ bdw, const float* __restrict__ wo,
                const float* __restrict__ bo,  float* __restrict__ out)
{
    extern __shared__ char smem[];
    _Float16* xs   = (_Float16*)(smem + B_XS);    // [336][72]  x halo tile [pix][chan]
    _Float16* whs  = (_Float16*)(smem + B_WHS);   // [96][72]   Wh chunk (q,k,v rows)
    _Float16* hs   = (_Float16*)(smem + B_HS);    // [96][344]  hidden chunk [chan][halopix]
    float*    qs   = (float*)   (smem + B_QS);    // [32][264]  f32
    float*    ks   = (float*)   (smem + B_KS);    // [32][264]  f32
    _Float16* vs   = (_Float16*)(smem + B_VS);    // [32][264]
    _Float16* ts   = (_Float16*)(smem + B_TS);    // [256][40]  (v*conv) [pix][chan]
    _Float16* wos  = (_Float16*)(smem + B_WOS);   // [64][40]
    float*    wdws = (float*)   (smem + B_WDWS);  // [96][12]
    float*    bdws = (float*)   (smem + B_BDWS);  // [96]
    float*    bhs  = (float*)   (smem + B_BHS);   // [96]

    const int tid  = threadIdx.x;
    const int lane = tid & 31;
    const int wid  = tid >> 5;
    const int l15  = lane & 15;
    const int lhi  = lane >> 4;

    const int b  = blockIdx.z;
    const int y0 = blockIdx.y * TILE;
    const int x0 = blockIdx.x * TILE;

    // ---- stage 0: x halo tile (18x18x64) -> LDS f16, [pixel][chan] ----
    for (int idx = tid; idx < DIM * HP; idx += 256) {
        int c  = idx / HP;
        int hp = idx - c * HP;
        int gy = y0 - 1 + hp / HALO;
        int gx = x0 - 1 + hp % HALO;
        float v = 0.f;
        if (gy >= 0 && gy < HH && gx >= 0 && gx < WW)
            v = x[((b * DIM + c) * HH + gy) * WW + gx];
        xs[hp * XS_STRIDE + c] = (_Float16)v;
    }

    // GEMM2 accumulators: out[64][256] as 4x16 tiles of 16x16; 8 tiles per wave.
    v8f acc[8];
#pragma unroll
    for (int t = 0; t < 8; ++t)
#pragma unroll
        for (int i = 0; i < 8; ++i) acc[t][i] = 0.f;

    const int px_t = tid & 15;          // this thread's tile pixel (stages 3)
    const int py_t = tid >> 4;

    for (int chunk = 0; chunk < 4; ++chunk) {
        const int qc0 = chunk * 32;
        __syncthreads();   // previous chunk's GEMM2 reads of ts/wos done

        // ---- stage 1: per-chunk weights -> LDS ----
        if (chunk < 3) __builtin_prefetch(&wh[(qc0 + 32) * DIM], 0, 1);
        for (int idx = tid; idx < 96 * 64; idx += 256) {
            int r = idx >> 6, c = idx & 63;
            int gch = (r >> 5) * C2 + qc0 + (r & 31);
            whs[r * WH_STRIDE + c] = (_Float16)wh[gch * DIM + c];
        }
        for (int idx = tid; idx < 96 * 9; idx += 256) {
            int r = idx / 9, c = idx - r * 9;
            int gch = (r >> 5) * C2 + qc0 + (r & 31);
            wdws[r * 12 + c] = wdw[gch * 9 + c];
        }
        for (int idx = tid; idx < 96; idx += 256) {
            int gch = (idx >> 5) * C2 + qc0 + (idx & 31);
            bdws[idx] = bdw[gch];
            bhs[idx]  = bh[gch];
        }
        for (int idx = tid; idx < 64 * 32; idx += 256) {
            int r = idx >> 5, c = idx & 31;
            wos[r * WO_STRIDE + c] = (_Float16)wo[r * C2 + qc0 + c];
        }
        __syncthreads();

        // ---- stage 2: GEMM1 (WMMA): hidden[96][336] = Wh_chunk @ x_halo ----
        for (int t = wid; t < 6 * 21; t += 8) {
            int rt = t / 21, ct = t - rt * 21;
            v8f c8;
#pragma unroll
            for (int i = 0; i < 8; ++i) c8[i] = 0.f;
#pragma unroll
            for (int k0 = 0; k0 < 64; k0 += 32) {
                int arow  = rt * 16 + l15;
                int baseK = lhi ? 8 : 0;
                v8h alo = *(const v8h*)&whs[arow * WH_STRIDE + k0 + baseK];
                v8h ahi = *(const v8h*)&whs[arow * WH_STRIDE + k0 + baseK + 16];
                v16h A;
#pragma unroll
                for (int i = 0; i < 8; ++i) { A[i] = alo[i]; A[8 + i] = ahi[i]; }
                int hp = ct * 16 + l15;
                int kb = k0 + (lhi ? 8 : 0);
                v8h blo = *(const v8h*)&xs[hp * XS_STRIDE + kb];
                v8h bhi = *(const v8h*)&xs[hp * XS_STRIDE + kb + 16];
                v16h B;
#pragma unroll
                for (int i = 0; i < 8; ++i) { B[i] = blo[i]; B[8 + i] = bhi[i]; }
                c8 = __builtin_amdgcn_wmma_f32_16x16x32_f16(
                        false, A, false, B, (short)0, c8, false, false);
            }
            // branch-free masked store (+hidden bias); rows padded so hp<336 is safe
            int hp = ct * 16 + l15;
            int gy = y0 - 1 + hp / HALO;
            int gx = x0 - 1 + hp % HALO;
            float mask = (hp < HP && gy >= 0 && gy < HH && gx >= 0 && gx < WW) ? 1.f : 0.f;
#pragma unroll
            for (int i = 0; i < 8; ++i) {
                int chl = rt * 16 + i + (lhi ? 8 : 0);
                float v = (c8[i] + bhs[chl]) * mask;
                hs[chl * HS_STRIDE + hp] = (_Float16)v;
            }
        }
        __syncthreads();

        // ---- stage 3: 3x3 depthwise conv; each thread owns one tile pixel ----
        {
            const int hb = (py_t) * HALO + px_t;   // halo base for (py,px): offsets +{0..2}
#pragma unroll 1
            for (int c3 = 0; c3 < 32; ++c3) {      // q channels
                const float* wp = &wdws[c3 * 12];
                const _Float16* hr = &hs[c3 * HS_STRIDE + hb];
                float a = bdws[c3];
#pragma unroll
                for (int dy = 0; dy < 3; ++dy)
#pragma unroll
                    for (int dx = 0; dx < 3; ++dx)
                        a += wp[dy * 3 + dx] * (float)hr[dy * HALO + dx];
                qs[c3 * QKV_STRIDE + tid] = a;
            }
#pragma unroll 1
            for (int c3 = 0; c3 < 32; ++c3) {      // k channels
                const float* wp = &wdws[(32 + c3) * 12];
                const _Float16* hr = &hs[(32 + c3) * HS_STRIDE + hb];
                float a = bdws[32 + c3];
#pragma unroll
                for (int dy = 0; dy < 3; ++dy)
#pragma unroll
                    for (int dx = 0; dx < 3; ++dx)
                        a += wp[dy * 3 + dx] * (float)hr[dy * HALO + dx];
                ks[c3 * QKV_STRIDE + tid] = a;
            }
#pragma unroll 1
            for (int c3 = 0; c3 < 32; ++c3) {      // v channels
                const float* wp = &wdws[(64 + c3) * 12];
                const _Float16* hr = &hs[(64 + c3) * HS_STRIDE + hb];
                float a = bdws[64 + c3];
#pragma unroll
                for (int dy = 0; dy < 3; ++dy)
#pragma unroll
                    for (int dx = 0; dx < 3; ++dx)
                        a += wp[dy * 3 + dx] * (float)hr[dy * HALO + dx];
                vs[c3 * VS_STRIDE + tid] = (_Float16)a;
            }
        }
        __syncthreads();

        // ---- stage 4: 8x8 circular conv q(*)k, times v.  Unit = (ch,patch,row u).
        // All rotations inside are compile-time constants -> pure v_fma_f32.
#pragma unroll 1
        for (int t = 0; t < 4; ++t) {
            int unit = tid + t * 256;              // 0..1023
            int ch = unit >> 5;                    // 0..31
            int pY = (unit >> 4) & 1;
            int pX = (unit >> 3) & 1;
            int uu = unit & 7;
            const float* qb = &qs[ch * QKV_STRIDE + pY * 128 + pX * 8];
            const float* kb = &ks[ch * QKV_STRIDE + pY * 128 + pX * 8];
            float o[8];
#pragma unroll
            for (int v = 0; v < 8; ++v) o[v] = 0.f;
#pragma unroll 2
            for (int i = 0; i < 8; ++i) {
                v4f q0 = *(const v4f*)&qb[i * 16];
                v4f q1 = *(const v4f*)&qb[i * 16 + 4];
                int kr = (uu - i) & 7;
                v4f k0 = *(const v4f*)&kb[kr * 16];
                v4f k1 = *(const v4f*)&kb[kr * 16 + 4];
                float qv[8] = {q0[0],q0[1],q0[2],q0[3],q1[0],q1[1],q1[2],q1[3]};
                float kv[8] = {k0[0],k0[1],k0[2],k0[3],k1[0],k1[1],k1[2],k1[3]};
#pragma unroll
                for (int v = 0; v < 8; ++v)
#pragma unroll
                    for (int j = 0; j < 8; ++j)
                        o[v] += qv[j] * kv[(v - j) & 7];
            }
            int prow = pY * 8 + uu;
#pragma unroll
            for (int v = 0; v < 8; ++v) {
                int p = prow * 16 + pX * 8 + v;
                float tv = o[v] * (float)vs[ch * VS_STRIDE + p];
                ts[p * TS_STRIDE + ch] = (_Float16)tv;   // [pix][chan]
            }
        }
        __syncthreads();

        // ---- stage 5: GEMM2 (WMMA) accumulate: out += Wout[:,chunk] @ t ----
#pragma unroll
        for (int t8 = 0; t8 < 8; ++t8) {
            int t  = wid * 8 + t8;
            int rt = t >> 4, ct = t & 15;
            int arow  = rt * 16 + l15;
            int baseK = lhi ? 8 : 0;
            v8h alo = *(const v8h*)&wos[arow * WO_STRIDE + baseK];
            v8h ahi = *(const v8h*)&wos[arow * WO_STRIDE + baseK + 16];
            v16h A;
#pragma unroll
            for (int i = 0; i < 8; ++i) { A[i] = alo[i]; A[8 + i] = ahi[i]; }
            int p  = ct * 16 + l15;
            int kb = lhi ? 8 : 0;
            v8h blo = *(const v8h*)&ts[p * TS_STRIDE + kb];
            v8h bhi = *(const v8h*)&ts[p * TS_STRIDE + kb + 16];
            v16h B;
#pragma unroll
            for (int i = 0; i < 8; ++i) { B[i] = blo[i]; B[8 + i] = bhi[i]; }
            acc[t8] = __builtin_amdgcn_wmma_f32_16x16x32_f16(
                          false, A, false, B, (short)0, acc[t8], false, false);
        }
    }

    // ---- stage 6: add output bias, store ----
#pragma unroll
    for (int t8 = 0; t8 < 8; ++t8) {
        int t  = wid * 8 + t8;
        int rt = t >> 4, ct = t & 15;
        int gy = y0 + ct;
        int gx = x0 + l15;
#pragma unroll
        for (int i = 0; i < 8; ++i) {
            int ch = rt * 16 + i + (lhi ? 8 : 0);
            out[((b * DIM + ch) * HH + gy) * WW + gx] = acc[t8][i] + bo[ch];
        }
    }
}

extern "C" void kernel_launch(void* const* d_in, const int* in_sizes, int n_in,
                              void* d_out, int out_size, void* d_ws, size_t ws_size,
                              hipStream_t stream) {
    const float* x   = (const float*)d_in[0];
    const float* wh  = (const float*)d_in[1];
    const float* bh  = (const float*)d_in[2];
    const float* wdw = (const float*)d_in[3];
    const float* bdw = (const float*)d_in[4];
    const float* wo  = (const float*)d_in[5];
    const float* bo  = (const float*)d_in[6];
    float* outp = (float*)d_out;
    (void)in_sizes; (void)n_in; (void)out_size; (void)d_ws; (void)ws_size;

    dim3 grid(WW / TILE, HH / TILE, 4);
    hipLaunchKernelGGL(fsas_fused, grid, dim3(256), SMEM_BYTES, stream,
                       x, wh, bh, wdw, bdw, wo, bo, outp);
}
